// Unidir_LSTM_46651934769703
// MI455X (gfx1250) — matrix operand order repaired and checked by reading, not compiled, not run on hardware
//
#include <hip/hip_runtime.h>
#include <stdint.h>

typedef __attribute__((ext_vector_type(16))) _Float16 v16h;
typedef __attribute__((ext_vector_type(8)))  _Float16 v8h;
typedef __attribute__((ext_vector_type(8)))  float    v8f;

#define E_SZ 512
#define H_SZ 1024
#define B_SZ 128
#define S_SZ 256
#define G3   3072   // 3*H: gates i, g, o (f gate is dead in the reference)
#define KCH  64     // K chunk staged in LDS per iteration
#define APAD 72     // padded LDS row stride (elements): 144 B, 16B-aligned, bank-spread

// ---------------------------------------------------------------------------
// One-time prep: repack W_ih/W_hh rows {i,g,o} to f16, fold biases.
// ---------------------------------------------------------------------------
__global__ __launch_bounds__(256) void prep_weights(
    const float* __restrict__ W_ih, const float* __restrict__ W_hh,
    const float* __restrict__ b_ih, const float* __restrict__ b_hh,
    _Float16* __restrict__ Wihb, _Float16* __restrict__ Whhb,
    float* __restrict__ bias3)
{
    const int n    = blockIdx.x;                       // 0..3071
    const int wrow = n + (n >= H_SZ ? H_SZ : 0);       // skip dead f-gate rows
    const float*  si = W_ih + (size_t)wrow * E_SZ;
    _Float16*     di = Wihb + (size_t)n    * E_SZ;
    for (int e = threadIdx.x; e < E_SZ; e += 256) di[e] = (_Float16)si[e];
    const float*  sh = W_hh + (size_t)wrow * H_SZ;
    _Float16*     dh = Whhb + (size_t)n    * H_SZ;
    for (int k = threadIdx.x; k < H_SZ; k += 256) dh[k] = (_Float16)sh[k];
    if (threadIdx.x == 0) bias3[n] = b_ih[wrow] + b_hh[wrow];
}

// ---------------------------------------------------------------------------
// One-time prep: embedding gather -> f16 X buffer, layout [S][B][E] row-major.
// ---------------------------------------------------------------------------
__global__ __launch_bounds__(256) void prep_embed(
    const int* __restrict__ inputs, const float* __restrict__ table,
    _Float16* __restrict__ Xb)
{
    const int sb = blockIdx.x;            // s*B + b
    const int s  = sb >> 7;               // B = 128
    const int b  = sb & 127;
    const int tok = inputs[b * S_SZ + s]; // inputs is [B, S]
    const float* src = table + (size_t)tok * E_SZ;
    _Float16*    dst = Xb    + (size_t)sb  * E_SZ;
    for (int e = threadIdx.x; e < E_SZ; e += 256)
        dst[e] = (_Float16)src[e];
}

__global__ __launch_bounds__(256) void zero_h(_Float16* __restrict__ h)
{
    h[blockIdx.x * 256 + threadIdx.x] = (_Float16)0.0f;
}

// ---------------------------------------------------------------------------
// Async global->LDS DMA of one 16B granule per lane (CDNA5, ASYNCcnt-tracked).
// GVS mode: 64-bit SGPR base + per-lane 32-bit byte offset.
// ---------------------------------------------------------------------------
__device__ __forceinline__ void async_b128(unsigned lds_byte_addr,
                                           const void* base,
                                           unsigned glob_byte_off)
{
    asm volatile("global_load_async_to_lds_b128 %0, %1, %2"
                 :: "v"(lds_byte_addr), "v"(glob_byte_off),
                    "s"((unsigned long long)(uintptr_t)base)
                 : "memory");
}

__device__ __forceinline__ void wait_async0()
{
    asm volatile("s_wait_asynccnt 0x0" ::: "memory");
}

// Stage one K chunk: A[128 x 64] (4 granules/thread) + B[64 x 64] (2/thread).
__device__ __forceinline__ void stage_chunk(
    int t, int nb, const _Float16* aSrc, const _Float16* bSrc,
    int lda, int k0, _Float16* ldsA, _Float16* ldsB)
{
#pragma unroll
    for (int i = 0; i < 4; ++i) {
        const int g   = t + (i << 8);
        const int row = g >> 3;
        const int gk  = (g & 7) << 3;
        async_b128((unsigned)(uintptr_t)&ldsA[row * APAD + gk], aSrc,
                   (unsigned)(((unsigned)row * lda + k0 + gk) * 2u));
    }
#pragma unroll
    for (int i = 0; i < 2; ++i) {
        const int g   = t + (i << 8);
        const int col = g >> 3;
        const int gk  = (g & 7) << 3;
        async_b128((unsigned)(uintptr_t)&ldsB[col * APAD + gk], bSrc,
                   (unsigned)(((unsigned)(nb + col) * lda + k0 + gk) * 2u));
    }
}

// ---------------------------------------------------------------------------
// Per-step GEMM: gates[B,3H] = X_s[B,E] @ Wihb^T + Hp[B,H] @ Whhb^T + bias3.
// Block tile M=128 x N=64 (48 blocks), 8 waves, double-buffered async LDS
// staging (one barrier per K chunk), fragments via ds_load -> v_wmma f16.
// ---------------------------------------------------------------------------
__global__ __launch_bounds__(256) void lstm_gates_wmma(
    const _Float16* __restrict__ X,     // [B,E]
    const _Float16* __restrict__ Wih,   // [3H,E]
    const _Float16* __restrict__ Whh,   // [3H,H]
    const _Float16* __restrict__ Hp,    // [B,H]
    const float*    __restrict__ bias3, // [3H]
    float*          __restrict__ gates) // [B,3H]
{
    __shared__ _Float16 ldsA[2][B_SZ * APAD];  // 2 x 18,432 B
    __shared__ _Float16 ldsB[2][64   * APAD];  // 2 x  9,216 B

    const int t    = threadIdx.x;
    const int lane = t & 31;
    const int wave = t >> 5;
    const int l16  = lane & 15;
    const int half = lane >> 4;          // K-segment selector
    const int nb   = blockIdx.x << 6;    // N base (64 cols/block)
    const int mrow = (wave << 4) + l16;  // A row handled by this lane

    v8f acc[4];
#pragma unroll
    for (int j = 0; j < 4; ++j) {
        const float bv = bias3[nb + j * 16 + l16];
#pragma unroll
        for (int r = 0; r < 8; ++r) acc[j][r] = bv;
    }

    // Unified K space: chunks 0..7 -> X / W_ih (K=512), 8..23 -> Hp / W_hh.
    // Prologue: stage chunk 0 into buffer 0.
    stage_chunk(t, nb, X, Wih, E_SZ, 0, ldsA[0], ldsB[0]);

    for (int ch = 0; ch < 24; ++ch) {
        wait_async0();      // my chunk-ch granules have landed in LDS
        __syncthreads();    // everyone's landed; everyone done with ch-1

        if (ch + 1 < 24) {  // overlap next chunk's DMA with this chunk's math
            const int cn = ch + 1;
            const _Float16* aS;
            const _Float16* bS;
            int lda, k0;
            if (cn < 8) { aS = X;  bS = Wih; lda = E_SZ; k0 = cn << 6; }
            else        { aS = Hp; bS = Whh; lda = H_SZ; k0 = (cn - 8) << 6; }
            stage_chunk(t, nb, aS, bS, lda, k0, ldsA[cn & 1], ldsB[cn & 1]);
        }

        // ---- consume buffer ch&1: 2 sub-chunks of K=32, 4 WMMAs each ----
        const _Float16* lA = ldsA[ch & 1];
        const _Float16* lB = ldsB[ch & 1];
#pragma unroll
        for (int ks = 0; ks < KCH; ks += 32) {
            const v8h lo = *(const v8h*)&lA[mrow * APAD + ks + half * 8];
            const v8h hi = *(const v8h*)&lA[mrow * APAD + ks + 16 + half * 8];
            v16h a;
#pragma unroll
            for (int i2 = 0; i2 < 8; ++i2) { a[i2] = lo[i2]; a[i2 + 8] = hi[i2]; }
#pragma unroll
            for (int j = 0; j < 4; ++j) {
                const v8h blo = *(const v8h*)&lB[(j * 16 + l16) * APAD + ks + half * 8];
                const v8h bhi = *(const v8h*)&lB[(j * 16 + l16) * APAD + ks + 16 + half * 8];
                v16h bb;
#pragma unroll
                for (int i2 = 0; i2 < 8; ++i2) { bb[i2] = blo[i2]; bb[i2 + 8] = bhi[i2]; }
                acc[j] = __builtin_amdgcn_wmma_f32_16x16x32_f16(
                    false, a, false, bb, (short)0, acc[j], false, false);
            }
        }
    }

    // ---- epilogue: C/D layout -> row-major f32 gates ----
    const int m0 = (wave << 4) + (half << 3);
#pragma unroll
    for (int j = 0; j < 4; ++j) {
#pragma unroll
        for (int r = 0; r < 8; ++r) {
            gates[(size_t)(m0 + r) * G3 + nb + j * 16 + l16] = acc[j][r];
        }
    }
}

// ---------------------------------------------------------------------------
// Per-step pointwise: c = sig(i)*tanh(g); h = sig(o)*tanh(c).
// ---------------------------------------------------------------------------
__global__ __launch_bounds__(256) void lstm_pointwise(
    const float* __restrict__ gates, _Float16* __restrict__ hf16,
    float* __restrict__ hout, int write_out)
{
    const int idx = blockIdx.x * 256 + threadIdx.x;   // B*H threads
    const int b = idx >> 10;
    const int k = idx & (H_SZ - 1);
    const float* g = gates + (size_t)b * G3;
    const float ig = g[k];
    const float gg = g[H_SZ + k];
    const float og = g[2 * H_SZ + k];
    const float si = 1.0f / (1.0f + __expf(-ig));
    const float so = 1.0f / (1.0f + __expf(-og));
    const float c  = si * tanhf(gg);
    const float h  = so * tanhf(c);
    hf16[idx] = (_Float16)h;
    if (write_out) hout[idx] = h;
}

// ---------------------------------------------------------------------------
extern "C" void kernel_launch(void* const* d_in, const int* in_sizes, int n_in,
                              void* d_out, int out_size, void* d_ws, size_t ws_size,
                              hipStream_t stream) {
    const int*   inputs = (const int*)  d_in[0];   // [B,S] int32
    const float* table  = (const float*)d_in[1];   // [V,E]
    const float* W_ih   = (const float*)d_in[2];   // [4H,E]
    const float* W_hh   = (const float*)d_in[3];   // [4H,H]
    const float* b_ih   = (const float*)d_in[4];   // [4H]
    const float* b_hh   = (const float*)d_in[5];   // [4H]
    float* out = (float*)d_out;                    // [B,H]

    // Workspace carve-up (all offsets 256B-aligned), ~44.8 MB total.
    char* ws = (char*)d_ws;
    _Float16* Xb    = (_Float16*)(ws + 0);         // S*B*E f16   = 33,554,432 B
    _Float16* Wihb  = (_Float16*)(ws + 33554432);  // 3H*E  f16   =  3,145,728 B
    _Float16* Whhb  = (_Float16*)(ws + 36700160);  // 3H*H  f16   =  6,291,456 B
    float*    bias3 = (float*)   (ws + 42991616);  // 3H    f32   =     12,288 B
    _Float16* hf16  = (_Float16*)(ws + 43003904);  // B*H   f16   =    262,144 B
    float*    gates = (float*)   (ws + 43266048);  // B*3H  f32   =  1,572,864 B

    prep_weights<<<G3, 256, 0, stream>>>(W_ih, W_hh, b_ih, b_hh, Wihb, Whhb, bias3);
    prep_embed<<<S_SZ * B_SZ, 256, 0, stream>>>(inputs, table, Xb);
    zero_h<<<(B_SZ * H_SZ) / 256, 256, 0, stream>>>(hf16);

    for (int s = 0; s < S_SZ; ++s) {
        lstm_gates_wmma<<<G3 / 64, 256, 0, stream>>>(
            Xb + (size_t)s * B_SZ * E_SZ, Wihb, Whhb, hf16, bias3, gates);
        lstm_pointwise<<<(B_SZ * H_SZ) / 256, 256, 0, stream>>>(
            gates, hf16, out, s == S_SZ - 1 ? 1 : 0);
    }
}
